// PointRendModule_84155589198107
// MI455X (gfx1250) — matrix operand order repaired and checked by reading, not compiled, not run on hardware
//
#include <hip/hip_runtime.h>

typedef _Float16 f16;
typedef __attribute__((ext_vector_type(16))) _Float16 v16h;
typedef __attribute__((ext_vector_type(8)))  _Float16 v8h;
typedef __attribute__((ext_vector_type(8)))  float    v8f;

#define B_    16
#define CIN   256
#define P_    8096
#define HF    256
#define WF    256
#define HM    64
#define WM    64
#define FC    256
#define NP    48            // points per block (3 WMMA point-tiles)
#define KSTR  264           // padded LDS row stride in f16 (528B -> 4-bank rotate/row)
#define NBLK  ((P_ + NP - 1) / NP)   // 169

// ---------------------------------------------------------------------------
// Weight prepack: f32 row-major -> f16 in WMMA A-fragment lane order.
// packed index = ((T*8 + Kt)*32 + L)*16 + j   (T=oc tile, Kt=k tile, L=lane)
// lane<16 : row M=L,     K = j
// lane>=16: row M=L-16,  K = j + 16
// ---------------------------------------------------------------------------
__global__ void prepack_weights(const float* __restrict__ w1,
                                const float* __restrict__ w2,
                                const float* __restrict__ w3,
                                f16* __restrict__ packed) {
  int tid = blockIdx.x * blockDim.x + threadIdx.x;
  if (tid >= 3 * FC * FC) return;
  int m  = tid / (FC * FC);
  int r  = tid % (FC * FC);
  int j  = r & 15;
  int L  = (r >> 4) & 31;
  int Kt = (r >> 9) & 7;
  int T  = r >> 12;
  int oc = T * 16 + (L & 15);
  int k  = Kt * 32 + (j + ((L >= 16) ? 16 : 0));
  float v;
  if (m == 0)      v = w1[oc * 257 + k];   // fine part of w1 (K=0..255)
  else if (m == 1) v = w2[oc * 256 + k];
  else             v = w3[oc * 256 + k];
  packed[tid] = (f16)v;
}

// ---------------------------------------------------------------------------
// Fragment loaders
// ---------------------------------------------------------------------------
__device__ __forceinline__ v16h load_a_frag(const f16* __restrict__ wmat,
                                            int octile, int ktile, int lane) {
  const f16* p = wmat + ((((octile * 8 + ktile) * 32) + lane) << 4);
  return *(const v16h*)p;   // 32B contiguous, 32B aligned
}

// B fragment (32x16): lane n -> column (point) pt_base + (n&15),
// K = ktile*32 + (n>=16 ? 16 : 0) + [0..15], contiguous in LDS row.
__device__ __forceinline__ v16h load_b_frag(const f16* buf, int pt_base,
                                            int ktile, int lane) {
  const f16* p = buf + (pt_base + (lane & 15)) * KSTR + ktile * 32 + ((lane >> 4) << 4);
  v8h lo = *(const v8h*)p;
  v8h hi = *(const v8h*)(p + 8);
  return __builtin_shufflevector(lo, hi, 0, 1, 2, 3, 4, 5, 6, 7,
                                 8, 9, 10, 11, 12, 13, 14, 15);
}

// ---------------------------------------------------------------------------
// Bilinear sampling setup (matches reference: ix = x*W - 0.5, zero OOB taps)
// ---------------------------------------------------------------------------
__device__ __forceinline__ void bilin_setup(float px, float py, int H, int W,
                                            int* off, float* wgt) {
  float ix = px * (float)W - 0.5f;
  float iy = py * (float)H - 0.5f;
  float x0f = floorf(ix), y0f = floorf(iy);
  float fx = ix - x0f, fy = iy - y0f;
  int x0 = (int)x0f, y0 = (int)y0f;
  int x1 = x0 + 1,   y1 = y0 + 1;
  float vx0 = (x0 >= 0 && x0 < W) ? 1.f : 0.f;
  float vx1 = (x1 >= 0 && x1 < W) ? 1.f : 0.f;
  float vy0 = (y0 >= 0 && y0 < H) ? 1.f : 0.f;
  float vy1 = (y1 >= 0 && y1 < H) ? 1.f : 0.f;
  int x0c = min(max(x0, 0), W - 1), x1c = min(max(x1, 0), W - 1);
  int y0c = min(max(y0, 0), H - 1), y1c = min(max(y1, 0), H - 1);
  off[0] = y0c * W + x0c; off[1] = y0c * W + x1c;
  off[2] = y1c * W + x0c; off[3] = y1c * W + x1c;
  wgt[0] = (1.f - fy) * (1.f - fx) * vy0 * vx0;
  wgt[1] = (1.f - fy) * fx         * vy0 * vx1;
  wgt[2] = fy         * (1.f - fx) * vy1 * vx0;
  wgt[3] = fy         * fx         * vy1 * vx1;
}

// ---------------------------------------------------------------------------
// Fused: sample -> L1 -> L2 -> L3 -> final projection, 48 points / block
// ---------------------------------------------------------------------------
__launch_bounds__(256)
__global__ void pointrend_fused(const float* __restrict__ coarse,
                                const float* __restrict__ fine,
                                const float* __restrict__ coords,
                                const f16*   __restrict__ pw,
                                const float* __restrict__ w1raw,
                                const float* __restrict__ b1,
                                const float* __restrict__ b2,
                                const float* __restrict__ b3,
                                const float* __restrict__ wf,
                                const float* __restrict__ bf,
                                float* __restrict__ out) {
  __shared__ f16   bufA[NP * KSTR];
  __shared__ f16   bufB[NP * KSTR];
  __shared__ float coarse_s[NP];
  __shared__ int   s_off[NP][4];
  __shared__ float s_w[NP][4];

  const int b    = blockIdx.y;
  const int pt0  = blockIdx.x * NP;
  const int t    = threadIdx.x;
  const int lane = t & 31;
  const int wave = t >> 5;

  // ---- Stage 0a: per-point sample parameters + coarse sample -------------
  if (t < NP) {
    int p = pt0 + t; if (p > P_ - 1) p = P_ - 1;   // duplicate-clamp tail
    float px = coords[((size_t)b * P_ + p) * 2 + 0];
    float py = coords[((size_t)b * P_ + p) * 2 + 1];
    int offF[4]; float wgF[4];
    bilin_setup(px, py, HF, WF, offF, wgF);
    for (int i = 0; i < 4; ++i) { s_off[t][i] = offF[i]; s_w[t][i] = wgF[i]; }
    int offC[4]; float wgC[4];
    bilin_setup(px, py, HM, WM, offC, wgC);
    const float* ci = coarse + (size_t)b * HM * WM;
    coarse_s[t] = wgC[0] * ci[offC[0]] + wgC[1] * ci[offC[1]] +
                  wgC[2] * ci[offC[2]] + wgC[3] * ci[offC[3]];
  }
  __syncthreads();

  // ---- Stage 0b: gather fine features -> bufB (f16, point-major rows) ----
  {
    const float* img = fine + (size_t)b * CIN * HF * WF;
    int c = t;                                   // lane == channel per iter
    for (int pl = 0; pl < NP; ++pl) {
      const float* pc = img + (size_t)c * (HF * WF);
      float v = s_w[pl][0] * pc[s_off[pl][0]] + s_w[pl][1] * pc[s_off[pl][1]] +
                s_w[pl][2] * pc[s_off[pl][2]] + s_w[pl][3] * pc[s_off[pl][3]];
      bufB[pl * KSTR + c] = (f16)v;
    }
  }
  __syncthreads();

  // ---- MLP layers (WMMA f16 -> f32 accum) --------------------------------
  auto do_layer = [&](const f16* src, f16* dst, const f16* wmat,
                      const float* bias, bool add_coarse) {
    for (int job = wave; job < 16 * (NP / 16); job += 8) {   // 48 jobs, uniform/wave
      int T     = job / (NP / 16);   // oc tile 0..15
      int ptile = job % (NP / 16);   // point tile 0..2
      v8f acc = {};
      for (int kt = 0; kt < 8; ++kt) {
        v16h a  = load_a_frag(wmat, T, kt, lane);
        v16h bb = load_b_frag(src, ptile * 16, kt, lane);
        acc = __builtin_amdgcn_wmma_f32_16x16x32_f16(
            false, a, false, bb, (short)0, acc, false, false);
      }
      int n  = lane & 15;              // D: col = point
      int mh = (lane >> 4) * 8;        // D: row offset per lane half
      int pl = ptile * 16 + n;
      float cs = add_coarse ? coarse_s[pl] : 0.f;
      v8h hv;
      for (int r2 = 0; r2 < 8; ++r2) {
        int oc  = T * 16 + mh + r2;
        float v = acc[r2] + bias[oc];
        if (add_coarse) v += w1raw[oc * 257 + 256] * cs;  // coarse-logit column
        v = v > 0.f ? v : 0.f;
        hv[r2] = (f16)v;
      }
      *(v8h*)(dst + pl * KSTR + T * 16 + mh) = hv;   // one b128 LDS store
    }
    __syncthreads();
  };

  do_layer(bufB, bufA, pw,               b1, true);   // layer 1 (+coarse rank-1)
  do_layer(bufA, bufB, pw + FC * FC,     b2, false);  // layer 2
  do_layer(bufB, bufA, pw + 2 * FC * FC, b3, false);  // layer 3

  // ---- Final projection: out[p] = bf + wf . h3[p] ------------------------
  if (t < NP) {
    int p = pt0 + t;
    if (p < P_) {
      const v8h* hv = (const v8h*)(bufA + t * KSTR);
      float s = bf[0];
      for (int c8 = 0; c8 < FC / 8; ++c8) {
        v8h hh = hv[c8];
        for (int i = 0; i < 8; ++i) s += wf[c8 * 8 + i] * (float)hh[i];
      }
      out[(size_t)b * P_ + p] = s;
    }
  }
}

// ---------------------------------------------------------------------------
extern "C" void kernel_launch(void* const* d_in, const int* in_sizes, int n_in,
                              void* d_out, int out_size, void* d_ws, size_t ws_size,
                              hipStream_t stream) {
  const float* coarse = (const float*)d_in[0];
  const float* fine   = (const float*)d_in[1];
  const float* coords = (const float*)d_in[2];
  const float* w1 = (const float*)d_in[3];
  const float* b1 = (const float*)d_in[4];
  const float* w2 = (const float*)d_in[5];
  const float* b2 = (const float*)d_in[6];
  const float* w3 = (const float*)d_in[7];
  const float* b3 = (const float*)d_in[8];
  const float* wf = (const float*)d_in[9];
  const float* bf = (const float*)d_in[10];
  float* out  = (float*)d_out;
  f16*   pack = (f16*)d_ws;                 // 3 * 256*256 f16 = 384 KB

  prepack_weights<<<(3 * FC * FC + 255) / 256, 256, 0, stream>>>(w1, w2, w3, pack);

  dim3 grid(NBLK, B_);
  pointrend_fused<<<grid, 256, 0, stream>>>(coarse, fine, coords, pack, w1,
                                            b1, b2, b3, wf, bf, out);
}
